// HashGrid_78580721648016
// MI455X (gfx1250) — compile-verified, matrix operand order checked by reading, not run on hardware
//
#include <hip/hip_runtime.h>

typedef __attribute__((ext_vector_type(16))) _Float16 v16h;
typedef __attribute__((ext_vector_type(8)))  float    v8f;

namespace {
constexpr unsigned kT  = 1u << 19;      // table entries per level
constexpr unsigned kF  = 2;             // features per level
constexpr unsigned kP1 = 2654435761u;
constexpr unsigned kP2 = 805459861u;
// ceil(16 * (2048/16)^(l/15)) for l = 0..15
__device__ constexpr unsigned kRes[16] = {
    16, 23, 31, 43, 59, 81, 112, 154, 213, 295, 407, 562, 777, 1073, 1483, 2048
};
}

// 256 threads = 8 waves (wave32); each wave computes a 16-point tile fully
// (hash-grid encode -> WMMA layer1 -> relu -> WMMA layer2 -> sigma/geo).
__global__ __launch_bounds__(256)
void hashgrid_mlp_wmma(const float* __restrict__ xyzs,
                       const float* __restrict__ tables,   // [16][2^19][2] f32
                       const float* __restrict__ W1,       // [32][64] f32
                       const float* __restrict__ W2,       // [64][16] f32
                       float* __restrict__ out_sigma,      // [N]
                       float* __restrict__ out_geo,        // [N][15]
                       int npts)
{
    __shared__ _Float16 sW1[32 * 64];    // 4 KB, staged once per block
    __shared__ _Float16 sW2[64 * 16];    // 2 KB
    __shared__ _Float16 sH[8][16 * 64];  // per-wave hidden tile, 16 KB total

    const int tid = threadIdx.x;
    for (int i = tid; i < 32 * 64; i += 256) sW1[i] = (_Float16)W1[i];
    for (int i = tid; i < 64 * 16; i += 256) sW2[i] = (_Float16)W2[i];
    __syncthreads();

    const int lane = tid & 31;
    const int wave = tid >> 5;
    const int row  = lane & 15;   // point-in-tile (layer1 A row / D column)
    const int lh   = lane >> 4;   // lane half selects which K-block this lane owns
    const int tile = blockIdx.x * 8 + wave;

    const int p  = tile * 16 + row;
    const int pl = (p < npts) ? p : (npts - 1);   // clamp loads; stores guarded later

    // x01 = clip((x+1)/2, 0, 1)
    const float x01x = fminf(fmaxf((xyzs[pl * 3 + 0] + 1.0f) * 0.5f, 0.0f), 1.0f);
    const float x01y = fminf(fmaxf((xyzs[pl * 3 + 1] + 1.0f) * 0.5f, 0.0f), 1.0f);
    const float x01z = fminf(fmaxf((xyzs[pl * 3 + 2] + 1.0f) * 0.5f, 0.0f), 1.0f);

    // ---- Hash-grid encoding, laid out directly as the WMMA A fragment ----
    // A 16x32 f16: lanes 0-15 hold K {0-7,16-23}, lanes 16-31 hold K {8-15,24-31}
    // K = 2*level + feat  =>  lane half 0 owns levels {0-3, 8-11}, half 1 {4-7, 12-15}
    float ex[16];
#pragma unroll
    for (int i = 0; i < 8; ++i) {
        const int l0 = (i < 4) ? i : (i + 4);        // level if lane-half 0
        const unsigned r0 = kRes[l0];
        const unsigned r1 = kRes[l0 + 4];            // level if lane-half 1
        const unsigned res = lh ? r1 : r0;           // v_cndmask of two immediates
        const bool dense   = lh ? (r1 <= 79u) : (r0 <= 79u);  // (res+1)^3 <= 2^19
        const int  lvl     = l0 + lh * 4;
        const float* tbl = tables + (size_t)lvl * (size_t)(kT * kF);
        const unsigned s = res + 1u;

        const float px = x01x * (float)res;
        const float py = x01y * (float)res;
        const float pz = x01z * (float)res;
        const float fx = floorf(px), fy = floorf(py), fz = floorf(pz);
        const float wx = px - fx,   wy = py - fy,   wz = pz - fz;
        const unsigned ix = (unsigned)fx, iy = (unsigned)fy, iz = (unsigned)fz;

        float f0 = 0.0f, f1 = 0.0f;
#pragma unroll
        for (int c = 0; c < 8; ++c) {
            const unsigned cx = ix + (c & 1);
            const unsigned cy = iy + ((c >> 1) & 1);
            const unsigned cz = iz + (c >> 2);
            const unsigned di = cx + s * (cy + s * cz);
            const unsigned hi = (cx ^ (cy * kP1) ^ (cz * kP2)) & (kT - 1u);
            const unsigned idx = dense ? di : hi;
            const float w = ((c & 1)        ? wx : 1.0f - wx)
                          * (((c >> 1) & 1) ? wy : 1.0f - wy)
                          * ((c >> 2)       ? wz : 1.0f - wz);
            const float2 t = ((const float2*)tbl)[idx];   // L2-resident gather
            f0 = fmaf(t.x, w, f0);
            f1 = fmaf(t.y, w, f1);
        }
        ex[2 * i]     = f0;
        ex[2 * i + 1] = f1;
    }

    v16h a;
#pragma unroll
    for (int e = 0; e < 16; ++e) a[e] = (_Float16)ex[e];

    // ---- Layer 1: h[16x64] = enc[16x32] @ W1[32x64], 4 WMMAs ----
    // B 32x16 f16 (mirrors A): lane holds column n = g*16+row, K per lane-half.
    v8f acc[4];
#pragma unroll
    for (int g = 0; g < 4; ++g) {
        v16h b;
#pragma unroll
        for (int e = 0; e < 16; ++e) {
            const int k = ((e < 8) ? e : e + 8) + lh * 8;
            b[e] = sW1[k * 64 + g * 16 + row];
        }
        v8f z = {};
        acc[g] = __builtin_amdgcn_wmma_f32_16x16x32_f16(
            false, a, false, b, (short)0, z, false, false);
    }

    // ---- ReLU + transpose column-major D -> row-major A via LDS ----
    // D layout: lanes 0-15 VGPR r = (M=r, N=lane); lanes 16-31 = (M=r+8, N=lane-16)
    _Float16* myH = &sH[wave][0];
#pragma unroll
    for (int g = 0; g < 4; ++g) {
#pragma unroll
        for (int r = 0; r < 8; ++r) {
            const float v = acc[g][r];
            myH[(r + lh * 8) * 64 + g * 16 + row] = (_Float16)(v > 0.0f ? v : 0.0f);
        }
    }
    __syncthreads();

    // ---- Layer 2: out[16x16] = h[16x64] @ W2[64x16], K-chained WMMAs ----
    v16h ha, hb, wa, wb;
#pragma unroll
    for (int e = 0; e < 16; ++e) {
        const int k = ((e < 8) ? e : e + 8) + lh * 8;   // K within a 32-block
        ha[e] = myH[row * 64 + k];
        hb[e] = myH[row * 64 + 32 + k];
        wa[e] = sW2[k * 16 + row];
        wb[e] = sW2[(k + 32) * 16 + row];
    }
    v8f dz = {};
    v8f d = __builtin_amdgcn_wmma_f32_16x16x32_f16(
        false, ha, false, wa, (short)0, dz, false, false);
    d = __builtin_amdgcn_wmma_f32_16x16x32_f16(
        false, hb, false, wb, (short)0, d, false, false);

    // ---- Epilogue: lane `row` owns output feature `row` for 8 points ----
#pragma unroll
    for (int r = 0; r < 8; ++r) {
        const int m = r + lh * 8;
        const int q = tile * 16 + m;
        if (q < npts) {
            const float v = d[r];
            if (row == 0) out_sigma[q] = expf(v);
            else          out_geo[(size_t)q * 15 + (row - 1)] = v;
        }
    }
}

extern "C" void kernel_launch(void* const* d_in, const int* in_sizes, int n_in,
                              void* d_out, int out_size, void* d_ws, size_t ws_size,
                              hipStream_t stream) {
    const float* xyzs   = (const float*)d_in[0];
    const float* tables = (const float*)d_in[1];
    const float* W1     = (const float*)d_in[2];
    const float* W2     = (const float*)d_in[3];
    const int npts = in_sizes[0] / 3;

    float* out_sigma = (float*)d_out;
    float* out_geo   = out_sigma + npts;   // tuple outputs concatenated flat

    const int blocks = (npts + 127) / 128;  // 128 points per block (8 waves x 16)
    hashgrid_mlp_wmma<<<blocks, 256, 0, stream>>>(
        xyzs, tables, W1, W2, out_sigma, out_geo, npts);
}
